// MMGCN_29171417874439
// MI455X (gfx1250) — compile-verified, matrix operand order checked by reading, not run on hardware
//
#include <hip/hip_runtime.h>
#include <stdint.h>
#include <stddef.h>

// ---------------- types for gfx1250 WMMA ----------------
typedef __bf16 bf16_t;
typedef __attribute__((ext_vector_type(16))) __bf16 bf16x16;
typedef __attribute__((ext_vector_type(8)))  __bf16 bf16x8;
typedef __attribute__((ext_vector_type(8)))  float  f32x8;

#define NM_  16384
#define ND_  8192
#define FDIM 256
#define CHN  4
#define HIDN 20
#define OCN  128

#define KCH   64   // K-chunk staged in LDS per iteration
#define LDSW  72   // padded LDS row stride (bf16 elems): 72*2=144B => bank-conflict-free frags

union FragU { bf16x16 v; bf16x8 h[2]; };

__device__ __forceinline__ bf16x16 load_frag16(const bf16_t* p) {
  FragU u;
  u.h[0] = *(const bf16x8*)(p);        // K offsets +0..7 (relative)
  u.h[1] = *(const bf16x8*)(p + 16);   // K offsets +16..23 (relative)
  return u.v;
}

// ---------------- NT GEMM: C[M,N] = A[M,K](row) * B[N,K](row)^T (+bias, relu)
// Block = 8 waves (wave32). Each wave owns one 16-wide N tile; the block shares
// four stacked 16-high M tiles (64 rows) staged chunk-by-chunk in LDS.
// Fragment layout per CDNA5 ISA 16-bit matrix 16x32:
//   lane L: row = L&15, khalf = L>>4; elems 0..7 -> K = khalf*8 + 0..7,
//   elems 8..15 -> K = khalf*8 + 16..23.
__global__ void k_gemm_nt(const bf16_t* __restrict__ A,
                          const bf16_t* __restrict__ B,
                          float* __restrict__ C,
                          int M, int N, int K,
                          const float* __restrict__ bias, int do_relu)
{
  __shared__ __align__(16) bf16_t As[64 * LDSW];   // 9216 B

  const int lane = threadIdx.x & 31;
  const int wave = threadIdx.x >> 5;                 // 0..7
  const int tile_n  = (blockIdx.x * 8 + wave) * 16;
  const int tile_m0 = blockIdx.y * 64;
  const bool active = (tile_n < N);                  // grids are exact; stays true

  const int r  = lane & 15;
  const int kh = lane >> 4;                          // 0 or 1

  f32x8 zero = {0.f,0.f,0.f,0.f,0.f,0.f,0.f,0.f};
  f32x8 acc[4];
  acc[0] = zero; acc[1] = zero; acc[2] = zero; acc[3] = zero;

  const bf16_t* Bp = B + (size_t)(tile_n + r) * K + kh * 8;

  // staging map: 64 rows x 64 K = 512 sixteen-byte segments; 256 threads x 2 segs
  const int s0row = threadIdx.x >> 3;                // 0..31
  const int s0seg = threadIdx.x & 7;                 // 0..7

  for (int k0 = 0; k0 < K; k0 += KCH) {
    __syncthreads();
#pragma unroll
    for (int ss = 0; ss < 2; ++ss) {
      const int row = s0row + ss * 32;
      *(bf16x8*)&As[row * LDSW + s0seg * 8] =
          *(const bf16x8*)&A[(size_t)(tile_m0 + row) * K + k0 + s0seg * 8];
    }
    __syncthreads();

    if (active) {
#pragma unroll
      for (int j = 0; j < 2; ++j) {                  // two 16x16x32 steps per chunk
        bf16x16 bfrag = load_frag16(Bp + k0 + j * 32);
#pragma unroll
        for (int i = 0; i < 4; ++i) {
          bf16x16 afrag = load_frag16(&As[(i * 16 + r) * LDSW + j * 32 + kh * 8]);
          acc[i] = __builtin_amdgcn_wmma_f32_16x16x32_bf16(
              /*neg_a=*/false, afrag, /*neg_b=*/false, bfrag,
              /*c_mod=*/(short)0, acc[i], /*reuse_a=*/false, /*reuse_b=*/false);
        }
      }
    }
  }

  if (active) {
    // C/D layout: vgpr rr, lanes 0-15 -> M=rr, N=lane ; lanes 16-31 -> M=8+rr
    const int n = tile_n + r;
    const float bv = bias ? bias[n] : 0.f;
#pragma unroll
    for (int i = 0; i < 4; ++i) {
      const int mbase = tile_m0 + i * 16 + kh * 8;
#pragma unroll
      for (int rr = 0; rr < 8; ++rr) {
        float v = acc[i][rr] + bv;
        if (do_relu) v = fmaxf(v, 0.f);
        C[(size_t)(mbase + rr) * N + n] = v;
      }
    }
  }
}

// ---------------- helper kernels ----------------
__global__ void k_zero_f32(float* p, int n) {
  int i = blockIdx.x * blockDim.x + threadIdx.x;
  if (i < n) p[i] = 0.f;
}

__global__ void k_deg(float* deg, const int* __restrict__ dst,
                      const float* __restrict__ w, int E) {
  int e = blockIdx.x * blockDim.x + threadIdx.x;
  if (e < E) atomicAdd(&deg[dst[e]], w[e]);
}

// dinv = rsqrt(deg + 1)  (self-loop weight 1 => deg >= 1 always)
__global__ void k_rsqrt1(float* d, int n) {
  int i = blockIdx.x * blockDim.x + threadIdx.x;
  if (i < n) d[i] = rsqrtf(d[i] + 1.f);
}

__global__ void k_cvt_bf16(bf16_t* __restrict__ dst, const float* __restrict__ src, size_t n) {
  size_t i = (size_t)blockIdx.x * blockDim.x + threadIdx.x;
  if (i < n) dst[i] = (bf16_t)src[i];
}

// Wt[o*D + i] = bf16(W[i*D + o])  (transpose 256x256 weight for NT gemm)
__global__ void k_wt_bf16(bf16_t* __restrict__ Wt, const float* __restrict__ W, int D) {
  int i = threadIdx.x;   // input dim
  int o = blockIdx.x;    // output dim
  Wt[(size_t)o * D + i] = (bf16_t)W[(size_t)i * D + o];
}

// out[n,f] = bias[f]   (total = N*F, F power of two)
__global__ void k_init_bias(float* __restrict__ out, const float* __restrict__ b,
                            size_t total, int Fmask) {
  size_t i = (size_t)blockIdx.x * blockDim.x + threadIdx.x;
  if (i < total) out[i] = b[i & (size_t)Fmask];
}

// one block per edge; thread f handles feature f
__global__ void k_edge_scatter(float* __restrict__ out, const float* __restrict__ h,
                               const int* __restrict__ src, const int* __restrict__ dst,
                               const float* __restrict__ w, const float* __restrict__ dinv,
                               int F) {
  int e = blockIdx.x;
  int s = src[e], d = dst[e];
  float norm = dinv[s] * w[e] * dinv[d];
  const float* hs = h + (size_t)s * F;
  float* od = out + (size_t)d * F;
  for (int f = threadIdx.x; f < F; f += blockDim.x)
    atomicAdd(&od[f], norm * hs[f]);
}

// out = relu(out + dinv[n]^2 * h);  runs after scatter (stream-ordered, no atomics)
__global__ void k_self_relu(float* __restrict__ out, const float* __restrict__ h,
                            const float* __restrict__ dinv, size_t total) {
  size_t i = (size_t)blockIdx.x * blockDim.x + threadIdx.x;
  if (i >= total) return;
  int n = (int)(i >> 8);               // F == 256
  float di = dinv[n];
  out[i] = fmaxf(out[i] + di * di * h[i], 0.f);
}

// pack feat channel c into concat bf16 A [N, 4*256] and accumulate sum(feat) into s_acc[c]
__global__ void k_chan_pack(bf16_t* __restrict__ seA, const float* __restrict__ feat,
                            float* __restrict__ s_acc, int c, size_t total) {
  __shared__ float red[256];
  size_t i = (size_t)blockIdx.x * blockDim.x + threadIdx.x;
  float v = 0.f;
  if (i < total) {
    v = feat[i];
    size_t n = i >> 8;                 // F == 256
    int f = (int)(i & 255);
    seA[n * (CHN * FDIM) + (size_t)c * FDIM + f] = (bf16_t)v;
  }
  red[threadIdx.x] = v;
  __syncthreads();
  for (int st = 128; st > 0; st >>= 1) {
    if ((int)threadIdx.x < st) red[threadIdx.x] += red[threadIdx.x + st];
    __syncthreads();
  }
  if (threadIdx.x == 0) atomicAdd(&s_acc[c], red[0]);
}

// tiny SE attention: a = sigmoid(relu(mean @ fc1 + b1) @ fc2 + b2)
__global__ void k_se_attn(const float* __restrict__ s_sum, float inv_count,
                          const float* __restrict__ fc1w, const float* __restrict__ fc1b,
                          const float* __restrict__ fc2w, const float* __restrict__ fc2b,
                          float* __restrict__ a) {
  if (threadIdx.x != 0 || blockIdx.x != 0) return;
  float s[CHN];
  for (int c = 0; c < CHN; ++c) s[c] = s_sum[c] * inv_count;
  float hb[HIDN];
  for (int j = 0; j < HIDN; ++j) {
    float acc = fc1b[j];
    for (int c = 0; c < CHN; ++c) acc += s[c] * fc1w[c * HIDN + j];
    hb[j] = fmaxf(acc, 0.f);
  }
  for (int c = 0; c < CHN; ++c) {
    float acc = fc2b[c];
    for (int j = 0; j < HIDN; ++j) acc += hb[j] * fc2w[j * CHN + c];
    a[c] = 1.f / (1.f + expf(-acc));
  }
}

// Bse[o, c*256+f] = bf16(a[c] * cnn_w[o,c,f]); cnn_w is already [OC][CH][F] row-major
__global__ void k_scale_cnnw(bf16_t* __restrict__ Bse, const float* __restrict__ cw,
                             const float* __restrict__ a, int total) {
  int i = blockIdx.x * blockDim.x + threadIdx.x;
  if (i >= total) return;
  int c = (i >> 8) & 3;                // f fastest (256), then c (4), then o
  Bse[i] = (bf16_t)(a[c] * cw[i]);
}

// ---------------- host orchestration ----------------
static inline unsigned div_up(size_t a, unsigned b) { return (unsigned)((a + b - 1) / b); }

extern "C" void kernel_launch(void* const* d_in, const int* in_sizes, int n_in,
                              void* d_out, int out_size, void* d_ws, size_t ws_size,
                              hipStream_t stream) {
  (void)n_in; (void)out_size; (void)ws_size;

  const float* x_m       = (const float*)d_in[0];
  const float* x_d       = (const float*)d_in[1];
  const int*   mm_f_edges= (const int*)  d_in[2];
  const float* mm_f_w    = (const float*)d_in[3];
  const int*   mm_s_edges= (const int*)  d_in[4];
  const float* mm_s_w    = (const float*)d_in[5];
  const int*   dd_f_edges= (const int*)  d_in[6];
  const float* dd_f_w    = (const float*)d_in[7];
  const int*   dd_s_edges= (const int*)  d_in[8];
  const float* dd_s_w    = (const float*)d_in[9];
  // params, in dict insertion order:
  const float* W_x1f = (const float*)d_in[10]; const float* b_x1f = (const float*)d_in[11];
  const float* W_x2f = (const float*)d_in[12]; const float* b_x2f = (const float*)d_in[13];
  const float* W_x1s = (const float*)d_in[14]; const float* b_x1s = (const float*)d_in[15];
  const float* W_x2s = (const float*)d_in[16]; const float* b_x2s = (const float*)d_in[17];
  const float* W_y1f = (const float*)d_in[18]; const float* b_y1f = (const float*)d_in[19];
  const float* W_y2f = (const float*)d_in[20]; const float* b_y2f = (const float*)d_in[21];
  const float* W_y1s = (const float*)d_in[22]; const float* b_y1s = (const float*)d_in[23];
  const float* W_y2s = (const float*)d_in[24]; const float* b_y2s = (const float*)d_in[25];
  const float* fc1x_w = (const float*)d_in[26]; const float* fc1x_b = (const float*)d_in[27];
  const float* fc2x_w = (const float*)d_in[28]; const float* fc2x_b = (const float*)d_in[29];
  const float* fc1y_w = (const float*)d_in[30]; const float* fc1y_b = (const float*)d_in[31];
  const float* fc2y_w = (const float*)d_in[32]; const float* fc2y_b = (const float*)d_in[33];
  const float* cnnx_w = (const float*)d_in[34]; const float* cnnx_b = (const float*)d_in[35];
  const float* cnny_w = (const float*)d_in[36]; const float* cnny_b = (const float*)d_in[37];

  const int EM = in_sizes[3];
  const int ED = in_sizes[7];

  // ---- workspace carving (~100 MB) ----
  uint8_t* base = (uint8_t*)d_ws;
  size_t off = 0;
  auto carve = [&](size_t bytes) -> void* {
    off = (off + 255) & ~(size_t)255;
    void* p = base + off;
    off += bytes;
    return p;
  };
  float*  dinv_mf = (float*)carve((size_t)NM_ * 4);
  float*  dinv_ms = (float*)carve((size_t)NM_ * 4);
  float*  dinv_df = (float*)carve((size_t)ND_ * 4);
  float*  dinv_ds = (float*)carve((size_t)ND_ * 4);
  float*  hbuf    = (float*)carve((size_t)NM_ * FDIM * 4);   // 16 MB (disease uses prefix)
  float*  featA   = (float*)carve((size_t)NM_ * FDIM * 4);   // 16 MB
  float*  featB   = (float*)carve((size_t)NM_ * FDIM * 4);   // 16 MB
  bf16_t* convA   = (bf16_t*)carve((size_t)NM_ * FDIM * 2);  // 8 MB (also reused for final operands)
  bf16_t* Wt      = (bf16_t*)carve((size_t)FDIM * FDIM * 2);
  bf16_t* seA     = (bf16_t*)carve((size_t)NM_ * CHN * FDIM * 2); // 32 MB (reused per side)
  bf16_t* Bse     = (bf16_t*)carve((size_t)OCN * CHN * FDIM * 2);
  float*  x_out   = (float*)carve((size_t)NM_ * OCN * 4);    // 8 MB
  float*  y_out   = (float*)carve((size_t)ND_ * OCN * 4);    // 4 MB
  float*  s_acc   = (float*)carve(CHN * 4);
  float*  a_att   = (float*)carve(CHN * 4);
  bf16_t* Xo = convA;                          // 16384*128 bf16
  bf16_t* Yo = convA + (size_t)NM_ * OCN;      // 8192*128 bf16 (fits in convA's 8 MB)

  auto gemm = [&](const bf16_t* A, const bf16_t* B, float* C, int M, int N, int K,
                  const float* bias, int relu) {
    dim3 g((unsigned)(N / 128), (unsigned)(M / 64));
    k_gemm_nt<<<g, 256, 0, stream>>>(A, B, C, M, N, K, bias, relu);
  };

  auto degrees = [&](float* dinv, const int* edges, const float* w, int E, int Nn) {
    k_zero_f32<<<div_up(Nn, 256), 256, 0, stream>>>(dinv, Nn);
    k_deg<<<div_up(E, 256), 256, 0, stream>>>(dinv, edges + E, w, E);
    k_rsqrt1<<<div_up(Nn, 256), 256, 0, stream>>>(dinv, Nn);
  };

  auto conv = [&](const float* Xin, const float* W, const float* b,
                  const int* edges, const float* ew, const float* dinv,
                  int Nn, int E, float* outFeat) {
    size_t nt = (size_t)Nn * FDIM;
    k_cvt_bf16<<<div_up(nt, 256), 256, 0, stream>>>(convA, Xin, nt);
    k_wt_bf16<<<FDIM, FDIM, 0, stream>>>(Wt, W, FDIM);
    gemm(convA, Wt, hbuf, Nn, FDIM, FDIM, nullptr, 0);
    k_init_bias<<<div_up(nt, 256), 256, 0, stream>>>(outFeat, b, nt, FDIM - 1);
    k_edge_scatter<<<(unsigned)E, 256, 0, stream>>>(outFeat, hbuf, edges, edges + E, ew, dinv, FDIM);
    k_self_relu<<<div_up(nt, 256), 256, 0, stream>>>(outFeat, hbuf, dinv, nt);
  };

  auto pack = [&](const float* feat, int c, int Nn) {
    size_t nt = (size_t)Nn * FDIM;
    k_chan_pack<<<div_up(nt, 256), 256, 0, stream>>>(seA, feat, s_acc, c, nt);
  };

  // ---- degree / normalization (4 graphs) ----
  degrees(dinv_mf, mm_f_edges, mm_f_w, EM, NM_);
  degrees(dinv_ms, mm_s_edges, mm_s_w, EM, NM_);
  degrees(dinv_df, dd_f_edges, dd_f_w, ED, ND_);
  degrees(dinv_ds, dd_s_edges, dd_s_w, ED, ND_);

  // ---- miRNA side: 4 convs -> SE fuse -> x_out ----
  k_zero_f32<<<1, 32, 0, stream>>>(s_acc, CHN);
  conv(x_m,   W_x1f, b_x1f, mm_f_edges, mm_f_w, dinv_mf, NM_, EM, featA);
  pack(featA, 0, NM_);
  conv(featA, W_x2f, b_x2f, mm_f_edges, mm_f_w, dinv_mf, NM_, EM, featB);
  pack(featB, 1, NM_);
  conv(x_m,   W_x1s, b_x1s, mm_s_edges, mm_s_w, dinv_ms, NM_, EM, featA);
  pack(featA, 2, NM_);
  conv(featA, W_x2s, b_x2s, mm_s_edges, mm_s_w, dinv_ms, NM_, EM, featB);
  pack(featB, 3, NM_);
  k_se_attn<<<1, 32, 0, stream>>>(s_acc, 1.f / ((float)NM_ * FDIM),
                                  fc1x_w, fc1x_b, fc2x_w, fc2x_b, a_att);
  k_scale_cnnw<<<div_up(OCN * CHN * FDIM, 256), 256, 0, stream>>>(Bse, cnnx_w, a_att,
                                                                  OCN * CHN * FDIM);
  gemm(seA, Bse, x_out, NM_, OCN, CHN * FDIM, cnnx_b, 0);

  // ---- disease side ----
  k_zero_f32<<<1, 32, 0, stream>>>(s_acc, CHN);
  conv(x_d,   W_y1f, b_y1f, dd_f_edges, dd_f_w, dinv_df, ND_, ED, featA);
  pack(featA, 0, ND_);
  conv(featA, W_y2f, b_y2f, dd_f_edges, dd_f_w, dinv_df, ND_, ED, featB);
  pack(featB, 1, ND_);
  conv(x_d,   W_y1s, b_y1s, dd_s_edges, dd_s_w, dinv_ds, ND_, ED, featA);
  pack(featA, 2, ND_);
  conv(featA, W_y2s, b_y2s, dd_s_edges, dd_s_w, dinv_ds, ND_, ED, featB);
  pack(featB, 3, ND_);
  k_se_attn<<<1, 32, 0, stream>>>(s_acc, 1.f / ((float)ND_ * FDIM),
                                  fc1y_w, fc1y_b, fc2y_w, fc2y_b, a_att);
  k_scale_cnnw<<<div_up(OCN * CHN * FDIM, 256), 256, 0, stream>>>(Bse, cnny_w, a_att,
                                                                  OCN * CHN * FDIM);
  gemm(seA, Bse, y_out, ND_, OCN, CHN * FDIM, cnny_b, 0);

  // ---- final: out[16384, 8192] = x_out @ y_out^T  (NT gemm, K=128) ----
  k_cvt_bf16<<<div_up((size_t)NM_ * OCN, 256), 256, 0, stream>>>(Xo, x_out, (size_t)NM_ * OCN);
  k_cvt_bf16<<<div_up((size_t)ND_ * OCN, 256), 256, 0, stream>>>(Yo, y_out, (size_t)ND_ * OCN);
  gemm(Xo, Yo, (float*)d_out, NM_, ND_, OCN, nullptr, 0);
}
